// ChunkwiseDeltaAttention_24936580120990
// MI455X (gfx1250) — matrix operand order repaired and checked by reading, not compiled, not run on hardware
//
#include <hip/hip_runtime.h>
#include <hip/hip_bf16.h>

// ---------------------------------------------------------------------------
// ChunkwiseDeltaAttention for MI455X (gfx1250, wave32, WMMA bf16)
//
// Pipeline:
//   1) convert fp32 -> bf16 (hidden states + 5 weight matrices)
//   2) WMMA GEMM (double-buffered LDS, 256x128 tile, 64x64/wave): q,k,v,og
//   3) beta = softplus(hs @ b_w^T)          (tiny reduction kernel)
//   4) l2-normalize q,k rows of 128 (in-place, wave shuffle reduce)
//   5) gv = beta * v (in-place)
//   6) chunk attention: O = tril(Q K^T) @ GV  (two WMMA stages via LDS)
//   7) gated = attn * silu(og) (in-place)
//   8) WMMA GEMM: out = gated @ o_w^T + o_b  (f32 out)
// ---------------------------------------------------------------------------

typedef __attribute__((ext_vector_type(16))) __bf16 bf16x16;
typedef __attribute__((ext_vector_type(8)))  float  floatx8;

union FragBF { bf16x16 v; uint4 q[2]; };
union AccF   { floatx8 v; float f[8]; };

__device__ __forceinline__ float bf2f(unsigned short h) {
    union { unsigned u; float f; } x; x.u = ((unsigned)h) << 16; return x.f;
}
__device__ __forceinline__ unsigned short f2bf(float f) {
    union { float f; unsigned u; } x; x.f = f;
    unsigned r = x.u + 0x7fffu + ((x.u >> 16) & 1u);
    return (unsigned short)(r >> 16);
}

__device__ __forceinline__ floatx8 wmma_bf16(const FragBF& a, const FragBF& b, floatx8 c) {
    // D = A(16x32 bf16) * B(32x16 bf16) + C(16x16 f32)
    return __builtin_amdgcn_wmma_f32_16x16x32_bf16(
        false, a.v, false, b.v, (short)0, c, false, false);
}

// ---------------------------------------------------------------------------
// fp32 -> bf16 conversion
// ---------------------------------------------------------------------------
__global__ __launch_bounds__(256) void f32_to_bf16_kernel(
    const float* __restrict__ in, unsigned short* __restrict__ out, long long n) {
    long long i = (long long)blockIdx.x * blockDim.x + threadIdx.x;
    if (i < n) out[i] = f2bf(in[i]);
}

// ---------------------------------------------------------------------------
// GEMM: C[M,N] = A[M,K] @ W[N,K]^T (+bias)   (A,W bf16 row-major)
// block = 256 threads (8 waves), tile 256x128, K-step 32.
// Wave grid 4x2: each wave computes 64x64 = 4x4 tiles of 16x16 (16 WMMA/iter,
// 16 ds_load_b128/iter -> 1:1). Double-buffered LDS, one barrier per K-step.
// ---------------------------------------------------------------------------
#define GEMM_LDK 40   // 32 + 8 pad (elements)

__global__ __launch_bounds__(256) void gemm_bf16_kernel(
    const unsigned short* __restrict__ A,
    const unsigned short* __restrict__ W,
    const float* __restrict__ bias,     // may be null
    void* __restrict__ Cout,
    int M, int N, int K, int out_bf16) {

    __shared__ unsigned short lA[2][256 * GEMM_LDK];
    __shared__ unsigned short lB[2][128 * GEMM_LDK];

    const int tid  = threadIdx.x;
    const int lane = tid & 31;
    const int wave = tid >> 5;
    const int wm   = wave >> 1;          // 0..3 (64 rows each)
    const int wn   = wave & 1;           // 0..1 (64 cols each)
    const int half = lane >> 4;          // 0/1
    const int r    = lane & 15;

    const int rowA0 = blockIdx.y * 256;
    const int colB0 = blockIdx.x * 128;

    // staging map: A -> thread t covers row t, all 32 cols (4x uint4)
    //              B -> thread t covers row t>>1, 16 cols   (2x uint4)
    const unsigned short* gA = A + (size_t)(rowA0 + tid) * K;
    const unsigned short* gB = W + (size_t)(colB0 + (tid >> 1)) * K + (tid & 1) * 16;
    unsigned short* sArow = &lA[0][tid * GEMM_LDK];
    unsigned short* sBrow = &lB[0][(tid >> 1) * GEMM_LDK + (tid & 1) * 16];
    const size_t bufA = 256 * GEMM_LDK;   // element stride between buffers
    const size_t bufB = 128 * GEMM_LDK;

    AccF acc[4][4] = {};

    // prime buffer 0
    {
        uint4 a0 = *(const uint4*)(gA + 0);
        uint4 a1 = *(const uint4*)(gA + 8);
        uint4 a2 = *(const uint4*)(gA + 16);
        uint4 a3 = *(const uint4*)(gA + 24);
        uint4 b0 = *(const uint4*)(gB + 0);
        uint4 b1 = *(const uint4*)(gB + 8);
        *(uint4*)(sArow + 0)  = a0;
        *(uint4*)(sArow + 8)  = a1;
        *(uint4*)(sArow + 16) = a2;
        *(uint4*)(sArow + 24) = a3;
        *(uint4*)(sBrow + 0)  = b0;
        *(uint4*)(sBrow + 8)  = b1;
    }

    int cur = 0;
    for (int kb = 0; kb < K; kb += 32) {
        __syncthreads();

        const bool has_next = (kb + 32) < K;
        uint4 pa0, pa1, pa2, pa3, pb0, pb1;
        if (has_next) {
            const unsigned short* na = gA + kb + 32;
            const unsigned short* nb = gB + kb + 32;
            pa0 = *(const uint4*)(na + 0);
            pa1 = *(const uint4*)(na + 8);
            pa2 = *(const uint4*)(na + 16);
            pa3 = *(const uint4*)(na + 24);
            pb0 = *(const uint4*)(nb + 0);
            pb1 = *(const uint4*)(nb + 8);
        }

        const unsigned short* cA = &lA[0][cur * bufA];
        const unsigned short* cB = &lB[0][cur * bufB];

        // A fragments: lane pattern K = {half*8..+7, 16+half*8..+7}
        FragBF af[4];
#pragma unroll
        for (int mi = 0; mi < 4; ++mi) {
            const unsigned short* p = cA + (wm * 64 + mi * 16 + r) * GEMM_LDK + half * 8;
            af[mi].q[0] = *(const uint4*)p;
            af[mi].q[1] = *(const uint4*)(p + 16);
        }
#pragma unroll
        for (int ni = 0; ni < 4; ++ni) {
            // B fragment: lane pattern K = {half*8..+7, 16+half*8..+7}
            const unsigned short* p = cB + (wn * 64 + ni * 16 + r) * GEMM_LDK + half * 8;
            FragBF bf;
            bf.q[0] = *(const uint4*)p;
            bf.q[1] = *(const uint4*)(p + 16);
#pragma unroll
            for (int mi = 0; mi < 4; ++mi)
                acc[mi][ni].v = wmma_bf16(af[mi], bf, acc[mi][ni].v);
        }

        if (has_next) {
            const int nxt = cur ^ 1;
            unsigned short* dA = sArow + nxt * bufA;
            unsigned short* dB = sBrow + nxt * bufB;
            *(uint4*)(dA + 0)  = pa0;
            *(uint4*)(dA + 8)  = pa1;
            *(uint4*)(dA + 16) = pa2;
            *(uint4*)(dA + 24) = pa3;
            *(uint4*)(dB + 0)  = pb0;
            *(uint4*)(dB + 8)  = pb1;
        }
        cur ^= 1;
    }

    // epilogue: C layout lanes 0-15 -> M=vr, lanes 16-31 -> M=8+vr; N=lane&15
#pragma unroll
    for (int mi = 0; mi < 4; ++mi) {
#pragma unroll
        for (int ni = 0; ni < 4; ++ni) {
            const int col = colB0 + wn * 64 + ni * 16 + r;
            const float bv = bias ? bias[col] : 0.0f;
#pragma unroll
            for (int vr = 0; vr < 8; ++vr) {
                const int row = rowA0 + wm * 64 + mi * 16 + half * 8 + vr;
                const float val = acc[mi][ni].f[vr] + bv;
                const size_t off = (size_t)row * N + col;
                if (out_bf16) ((unsigned short*)Cout)[off] = f2bf(val);
                else          ((float*)Cout)[off] = val;
            }
        }
    }
}

// ---------------------------------------------------------------------------
// beta[t,h] = softplus(hs[t,:] . b_w[h,:] + b_b[h])   (fp32 inputs)
// one block per token, 256 threads = 16 heads x 16 partial lanes
// ---------------------------------------------------------------------------
__global__ __launch_bounds__(256) void beta_kernel(
    const float* __restrict__ hs, const float* __restrict__ bw,
    const float* __restrict__ bb, float* __restrict__ beta, int H) {
    __shared__ float red[256];
    const int tok = blockIdx.x;
    const int t = threadIdx.x;
    const int h = t >> 4;
    const int l = t & 15;
    const float* hr = hs + (size_t)tok * H;
    const float* wr = bw + (size_t)h * H;
    float s = 0.0f;
    for (int k = l; k < H; k += 16) s += hr[k] * wr[k];
    red[t] = s;
    __syncthreads();
    for (int off = 8; off > 0; off >>= 1) {
        if (l < off) red[t] += red[t + off];
        __syncthreads();
    }
    if (l == 0) {
        float x = red[t] + bb[h];
        float sp = (x > 20.0f) ? x : log1pf(expf(x));
        beta[(size_t)tok * 16 + h] = sp;
    }
}

// ---------------------------------------------------------------------------
// in-place L2 normalize rows of 128 bf16 (one wave per row)
// ---------------------------------------------------------------------------
__global__ __launch_bounds__(256) void l2norm_kernel(
    unsigned short* __restrict__ x, long long rows) {
    const long long wid = ((long long)blockIdx.x * blockDim.x + threadIdx.x) >> 5;
    const int lane = threadIdx.x & 31;
    if (wid >= rows) return;
    unsigned short* p = x + wid * 128 + lane * 4;
    uint2 u = *(const uint2*)p;
    float f0 = bf2f((unsigned short)(u.x & 0xffff));
    float f1 = bf2f((unsigned short)(u.x >> 16));
    float f2 = bf2f((unsigned short)(u.y & 0xffff));
    float f3 = bf2f((unsigned short)(u.y >> 16));
    float s = f0 * f0 + f1 * f1 + f2 * f2 + f3 * f3;
    for (int off = 16; off > 0; off >>= 1) s += __shfl_xor(s, off, 32);
    const float inv = 1.0f / fmaxf(sqrtf(s), 1e-12f);
    unsigned short o0 = f2bf(f0 * inv), o1 = f2bf(f1 * inv);
    unsigned short o2 = f2bf(f2 * inv), o3 = f2bf(f3 * inv);
    uint2 o;
    o.x = (unsigned)o0 | ((unsigned)o1 << 16);
    o.y = (unsigned)o2 | ((unsigned)o3 << 16);
    *(uint2*)p = o;
}

// ---------------------------------------------------------------------------
// gv = beta * v  (in-place on v, bf16)
// ---------------------------------------------------------------------------
__global__ __launch_bounds__(256) void gv_kernel(
    unsigned short* __restrict__ v, const float* __restrict__ beta, long long n) {
    long long i = (long long)blockIdx.x * blockDim.x + threadIdx.x;
    if (i >= n) return;
    const long long t = i >> 11;          // / 2048
    const int c = (int)(i & 2047);
    const int h = c >> 7;
    v[i] = f2bf(bf2f(v[i]) * beta[t * 16 + h]);
}

// ---------------------------------------------------------------------------
// gated = attn * silu(og)  (in-place on attn, bf16)
// ---------------------------------------------------------------------------
__global__ __launch_bounds__(256) void gate_kernel(
    unsigned short* __restrict__ attn, const unsigned short* __restrict__ og, long long n) {
    long long i = (long long)blockIdx.x * blockDim.x + threadIdx.x;
    if (i >= n) return;
    const float a = bf2f(attn[i]);
    const float g = bf2f(og[i]);
    const float sig = 1.0f / (1.0f + expf(-g));
    attn[i] = f2bf(a * g * sig);
}

// ---------------------------------------------------------------------------
// chunk attention: per (b, chunk, head):
//   S = Q(64x128) K^T (64x128)^T ; O = tril(S) @ GV(64x128)
// 128 threads = 4 waves, each wave owns a 16-row strip.
// K-chunk row-major and GV transposed staged in LDS; masked S
// round-trips through LDS to re-enter WMMA A-fragment layout.
// ---------------------------------------------------------------------------
#define KT_LD  136   // 128 + 8 pad
#define GV_LD  72    // 64 + 8 pad
#define SS_LD  72

__global__ __launch_bounds__(128) void attn_kernel(
    const unsigned short* __restrict__ qn,
    const unsigned short* __restrict__ kn,
    const unsigned short* __restrict__ gv,
    unsigned short* __restrict__ out,
    int L, int nchunks) {

    __shared__ unsigned short sK[64 * KT_LD];     // [j][d]
    __shared__ unsigned short sGvT[128 * GV_LD];  // [d][j]
    __shared__ unsigned short sS[64 * SS_LD];     // [i][j]

    const int blk = blockIdx.x;
    const int h = blk & 15;
    const int n = (blk >> 4) % nchunks;
    const int b = (blk >> 4) / nchunks;

    const int tid  = threadIdx.x;
    const int lane = tid & 31;
    const int wave = tid >> 5;          // 0..3
    const int half = lane >> 4;
    const int r    = lane & 15;

    const size_t tok0 = (size_t)b * L + (size_t)n * 64;

    // stage K chunk (row-major): 2 threads per row x 8 uint4
    {
        const int row = tid >> 1;
        const int ch  = (tid & 1) * 64;
        const unsigned short* g = kn + (tok0 + row) * 2048 + h * 128 + ch;
#pragma unroll
        for (int u = 0; u < 8; ++u) {
            uint4 x = *(const uint4*)(g + u * 8);
            *(uint4*)&sK[row * KT_LD + ch + u * 8] = x;
        }
    }
    // stage GV transposed: [d][j]
    for (int idx = tid; idx < 64 * 128; idx += 128) {
        const int j = idx >> 7;
        const int d = idx & 127;
        sGvT[d * GV_LD + j] = gv[(tok0 + j) * 2048 + h * 128 + d];
    }
    __syncthreads();

    const int i0 = wave * 16;

    // ---- phase 1: S strip (16 x 64) = Q strip @ K^T, K-dim = 128 ----
    AccF s[4] = {};
    const unsigned short* qrow = qn + (tok0 + i0 + r) * 2048 + h * 128;
#pragma unroll
    for (int kb = 0; kb < 128; kb += 32) {
        FragBF af;
        af.q[0] = *(const uint4*)(qrow + kb + half * 8);
        af.q[1] = *(const uint4*)(qrow + kb + 16 + half * 8);
#pragma unroll
        for (int nt = 0; nt < 4; ++nt) {
            const unsigned short* p = &sK[(nt * 16 + r) * KT_LD + kb + half * 8];
            FragBF bf;
            bf.q[0] = *(const uint4*)p;
            bf.q[1] = *(const uint4*)(p + 16);
            s[nt].v = wmma_bf16(af, bf, s[nt].v);
        }
    }

    // ---- mask: keep j <= i (decay factor is exp(0)=1 in-cone) ----
#pragma unroll
    for (int nt = 0; nt < 4; ++nt) {
        const int j = nt * 16 + r;
#pragma unroll
        for (int vr = 0; vr < 8; ++vr) {
            const int i = i0 + half * 8 + vr;
            const float v = (j <= i) ? s[nt].f[vr] : 0.0f;
            sS[i * SS_LD + j] = f2bf(v);
        }
    }
    // same-wave produce/consume of sS (DScnt handled by compiler waits)

    // ---- phase 2: O strip (16 x 128) = S strip @ GV, K-dim = 64 ----
    AccF o[8] = {};
#pragma unroll
    for (int kb = 0; kb < 64; kb += 32) {
        FragBF af;
        const unsigned short* p = &sS[(i0 + r) * SS_LD + kb + half * 8];
        af.q[0] = *(const uint4*)p;
        af.q[1] = *(const uint4*)(p + 16);
#pragma unroll
        for (int dt = 0; dt < 8; ++dt) {
            const unsigned short* pb = &sGvT[(dt * 16 + r) * GV_LD + kb + half * 8];
            FragBF bf;
            bf.q[0] = *(const uint4*)pb;
            bf.q[1] = *(const uint4*)(pb + 16);
            o[dt].v = wmma_bf16(af, bf, o[dt].v);
        }
    }

    // ---- store O strip ----
#pragma unroll
    for (int dt = 0; dt < 8; ++dt) {
        const int col = h * 128 + dt * 16 + r;
#pragma unroll
        for (int vr = 0; vr < 8; ++vr) {
            const int i = i0 + half * 8 + vr;
            out[(tok0 + i) * 2048 + col] = f2bf(o[dt].f[vr]);
        }
    }
}

// ---------------------------------------------------------------------------
// host: orchestrate the pipeline
// ---------------------------------------------------------------------------
extern "C" void kernel_launch(void* const* d_in, const int* in_sizes, int n_in,
                              void* d_out, int out_size, void* d_ws, size_t ws_size,
                              hipStream_t stream) {
    (void)in_sizes; (void)n_in; (void)out_size; (void)ws_size;

    const int B = 4, L = 4096, H = 2048, ND = 2048, NH = 16, CHUNK = 64;
    const long long TOK = (long long)B * L;           // 16384
    const long long TH  = TOK * H;                    // 33.5M
    const long long WW  = (long long)ND * H;          // 4.19M

    const float* hs   = (const float*)d_in[0];
    const float* q_w  = (const float*)d_in[1];
    const float* q_b  = (const float*)d_in[2];
    const float* k_w  = (const float*)d_in[3];
    const float* k_b  = (const float*)d_in[4];
    const float* v_w  = (const float*)d_in[5];
    const float* v_b  = (const float*)d_in[6];
    // d_in[7], d_in[8] = a_w, a_b (dead code in reference)
    const float* b_w  = (const float*)d_in[9];
    const float* b_b  = (const float*)d_in[10];
    const float* og_w = (const float*)d_in[11];
    const float* o_w  = (const float*)d_in[12];
    const float* o_b  = (const float*)d_in[13];

    unsigned short* w   = (unsigned short*)d_ws;
    unsigned short* hsb   = w;                 // TH
    unsigned short* qwB   = hsb + TH;          // WW
    unsigned short* kwB   = qwB + WW;
    unsigned short* vwB   = kwB + WW;
    unsigned short* ogwB  = vwB + WW;
    unsigned short* owB   = ogwB + WW;
    unsigned short* qb    = owB + WW;          // TH each
    unsigned short* kb2   = qb + TH;
    unsigned short* vb2   = kb2 + TH;
    unsigned short* ogb   = vb2 + TH;
    unsigned short* attnb = ogb + TH;
    float* beta = (float*)(attnb + TH);        // TOK*NH f32

    const dim3 blk256(256);

    // 1) conversions
    f32_to_bf16_kernel<<<dim3((unsigned)((TH + 255) / 256)), blk256, 0, stream>>>(hs, hsb, TH);
    f32_to_bf16_kernel<<<dim3((unsigned)((WW + 255) / 256)), blk256, 0, stream>>>(q_w, qwB, WW);
    f32_to_bf16_kernel<<<dim3((unsigned)((WW + 255) / 256)), blk256, 0, stream>>>(k_w, kwB, WW);
    f32_to_bf16_kernel<<<dim3((unsigned)((WW + 255) / 256)), blk256, 0, stream>>>(v_w, vwB, WW);
    f32_to_bf16_kernel<<<dim3((unsigned)((WW + 255) / 256)), blk256, 0, stream>>>(og_w, ogwB, WW);
    f32_to_bf16_kernel<<<dim3((unsigned)((WW + 255) / 256)), blk256, 0, stream>>>(o_w, owB, WW);

    // 2) projections (bf16 out): tile 256x128
    const dim3 ggrid(ND / 128, (unsigned)(TOK / 256));
    gemm_bf16_kernel<<<ggrid, blk256, 0, stream>>>(hsb, qwB, q_b, qb, (int)TOK, ND, H, 1);
    gemm_bf16_kernel<<<ggrid, blk256, 0, stream>>>(hsb, kwB, k_b, kb2, (int)TOK, ND, H, 1);
    gemm_bf16_kernel<<<ggrid, blk256, 0, stream>>>(hsb, vwB, v_b, vb2, (int)TOK, ND, H, 1);
    gemm_bf16_kernel<<<ggrid, blk256, 0, stream>>>(hsb, ogwB, nullptr, ogb, (int)TOK, ND, H, 1);

    // 3) beta
    beta_kernel<<<dim3((unsigned)TOK), blk256, 0, stream>>>(hs, b_w, b_b, beta, H);

    // 4) l2 normalize q, k (rows of 128)
    const long long rows = TOK * NH;
    l2norm_kernel<<<dim3((unsigned)(rows * 32 / 256)), blk256, 0, stream>>>(qb, rows);
    l2norm_kernel<<<dim3((unsigned)(rows * 32 / 256)), blk256, 0, stream>>>(kb2, rows);

    // 5) gv = beta * v
    gv_kernel<<<dim3((unsigned)((TH + 255) / 256)), blk256, 0, stream>>>(vb2, beta, TH);

    // 6) chunk attention
    const int nchunks = L / CHUNK;
    attn_kernel<<<dim3((unsigned)(B * nchunks * NH)), dim3(128), 0, stream>>>(
        qb, kb2, vb2, attnb, L, nchunks);

    // 7) output gate
    gate_kernel<<<dim3((unsigned)((TH + 255) / 256)), blk256, 0, stream>>>(attnb, ogb, TH);

    // 8) final projection (f32 out, + o_b)
    gemm_bf16_kernel<<<ggrid, blk256, 0, stream>>>(attnb, owB, o_b, (float*)d_out, (int)TOK, H, ND, 0);
}
